// Head_62251255988572
// MI455X (gfx1250) — matrix operand order repaired and checked by reading, compiled, not run on hardware
//
#include <hip/hip_runtime.h>

typedef __attribute__((ext_vector_type(16))) _Float16 v16h;
typedef __attribute__((ext_vector_type(8)))  _Float16 v8h;
typedef __attribute__((ext_vector_type(8)))  float    v8f;
typedef __attribute__((ext_vector_type(4)))  float    v4f;

#define B_ 4
#define T_ 4096
#define C_ 1024
#define H_ 64

// ---------------------------------------------------------------------------
// Kernel 0: WT[mat][h][c] = (f16) W[c][h]  (transpose+convert, 3*64*1024 elems)
// ---------------------------------------------------------------------------
__global__ __launch_bounds__(256) void wt_kernel(
    const float* __restrict__ Wq, const float* __restrict__ Wk,
    const float* __restrict__ Wv, _Float16* __restrict__ WT)
{
    const int idx = blockIdx.x * 256 + threadIdx.x;   // [3][64][1024]
    const int mat = idx >> 16;
    const int rem = idx & 65535;
    const int h   = rem >> 10;
    const int c   = rem & 1023;
    const float* W = (mat == 0) ? Wq : (mat == 1) ? Wk : Wv;
    WT[idx] = (_Float16)W[c * H_ + h];
}

// ---------------------------------------------------------------------------
// Kernel 1: fused q,k,v projection. One wave = one 16x16 row/col tile for ALL
// three matrices: A-fragment of x loaded+converted once, 3 WMMAs per k-step.
// q is pre-scaled by 1/sqrt(1024) so attention needs no scale.
// ---------------------------------------------------------------------------
__global__ __launch_bounds__(256) void qkv_proj_kernel(
    const float* __restrict__ x, const _Float16* __restrict__ WT,
    _Float16* __restrict__ qkv)   // [3][B*T][H]
{
    const int lane = threadIdx.x & 31;
    const int wave = threadIdx.x >> 5;
    const int task = blockIdx.x * 8 + wave;   // 1024 * 4 = 4096 tasks
    const int rowTile = task >> 2;
    const int colTile = task & 3;

    const int g   = lane >> 4;
    const int ln  = lane & 15;
    const int row = rowTile * 16 + ln;
    const int col = colTile * 16 + ln;

    v8f acc[3] = {};
    for (int k0 = 0; k0 < C_; k0 += 32) {
        const float* xrow = x + (size_t)row * C_ + k0;
        v4f a0 = *(const v4f*)(xrow + g * 8);
        v4f a1 = *(const v4f*)(xrow + g * 8 + 4);
        v4f a2 = *(const v4f*)(xrow + 16 + g * 8);
        v4f a3 = *(const v4f*)(xrow + 16 + g * 8 + 4);
        v16h a;
#pragma unroll
        for (int j = 0; j < 4; ++j) {
            a[j]      = (_Float16)a0[j];
            a[4 + j]  = (_Float16)a1[j];
            a[8 + j]  = (_Float16)a2[j];
            a[12 + j] = (_Float16)a3[j];
        }
#pragma unroll
        for (int mat = 0; mat < 3; ++mat) {
            const _Float16* wr = WT + mat * (H_ * C_) + col * C_ + k0 + g * 8;
            v8h lo = *(const v8h*)wr;
            v8h hi = *(const v8h*)(wr + 16);
            v16h bf;
#pragma unroll
            for (int j = 0; j < 8; ++j) { bf[j] = lo[j]; bf[8 + j] = hi[j]; }
            acc[mat] = __builtin_amdgcn_wmma_f32_16x16x32_f16(
                           false, a, false, bf, (short)0, acc[mat], false, false);
        }
    }
#pragma unroll
    for (int mat = 0; mat < 3; ++mat) {
        const float sc = (mat == 0) ? 0.03125f : 1.0f;   // fold 1/sqrt(C) into q
        _Float16* out = qkv + (size_t)mat * ((size_t)B_ * T_ * H_);
#pragma unroll
        for (int r = 0; r < 8; ++r) {
            const int m = r + 8 * g;
            out[(size_t)(rowTile * 16 + m) * H_ + col] = (_Float16)(acc[mat][r] * sc);
        }
    }
}

// ---------------------------------------------------------------------------
// Kernel 2: flash attention, double-buffered. Block = 128 threads (4 waves) =
// 64 q-rows of one batch; each wave owns 16 rows; s-blocks of 32.
// K staged by async global->LDS copies into ping-pong buffers; in-order
// ASYNCcnt completion lets "s_wait_asynccnt 2" retire the current block while
// the next block's copies stay in flight. V staged transposed via registers.
// One barrier per iteration (K copies go to buf[nxt]; V commit to buf[cur] before the barrier).
// ---------------------------------------------------------------------------
__global__ __launch_bounds__(128) void attn_kernel(
    const _Float16* __restrict__ q, const _Float16* __restrict__ k,
    const _Float16* __restrict__ v, float* __restrict__ out)
{
    __shared__ __attribute__((aligned(16))) _Float16 Kblk[2][32][H_];  // [buf][s][h]
    __shared__ __attribute__((aligned(16))) _Float16 VT[2][H_][32];    // [buf][h][s]
    __shared__ __attribute__((aligned(16))) _Float16 Pblk[4][16][32];

    const int lane = threadIdx.x & 31;
    const int wave = threadIdx.x >> 5;
    const int g  = lane >> 4;
    const int ln = lane & 15;

    const int b  = blockIdx.x / (T_ / 64);
    const int rb = blockIdx.x % (T_ / 64);
    const int rowbase = rb * 64;
    const int wrow = rowbase + wave * 16;

    const size_t baseBT = (size_t)b * T_ * H_;

    // Q fragments (h 0..31, 32..63), A layout, row = ln (q already has 1/32)
    v16h qf[2];
    {
        const _Float16* qrow = q + baseBT + (size_t)(wrow + ln) * H_;
#pragma unroll
        for (int hs = 0; hs < 2; ++hs) {
            v8h lo = *(const v8h*)(qrow + hs * 32 + g * 8);
            v8h hi = *(const v8h*)(qrow + hs * 32 + 16 + g * 8);
#pragma unroll
            for (int j = 0; j < 8; ++j) { qf[hs][j] = lo[j]; qf[hs][8 + j] = hi[j]; }
        }
    }

    float m_i[8], l_i[8];
    v8f o_acc[4] = {};
#pragma unroll
    for (int r = 0; r < 8; ++r) { m_i[r] = -__builtin_inff(); l_i[r] = 0.0f; }

    const int nblocks = (rowbase + 63) / 32 + 1;   // causal bound

    // staging addresses for this thread
    const int srow = threadIdx.x >> 2;          // 0..31
    const int scol = (threadIdx.x & 3) * 16;    // 0,16,32,48
    const _Float16* kbase = k + baseBT + (size_t)srow * H_ + scol;
    const _Float16* vbase = v + baseBT + (size_t)srow * H_ + scol;

    // prologue: async K block 0 -> buf 0; V block 0 -> registers
    v8h va, vb;
    {
        const unsigned ldsk = (unsigned)(uintptr_t)&Kblk[0][srow][scol];
        asm volatile("global_load_async_to_lds_b128 %0, %1, off"
                     :: "v"(ldsk), "v"(kbase) : "memory");
        asm volatile("global_load_async_to_lds_b128 %0, %1, off offset:16"
                     :: "v"(ldsk), "v"(kbase) : "memory");
        va = *(const v8h*)(vbase);
        vb = *(const v8h*)(vbase + 8);
    }

    for (int sb = 0; sb < nblocks; ++sb) {
        const int s0  = sb * 32;
        const int cur = sb & 1;
        const int nxt = cur ^ 1;

        // commit V registers -> VT[cur] (transposed scatter)
#pragma unroll
        for (int j = 0; j < 8; ++j) {
            VT[cur][scol + j][srow]     = va[j];
            VT[cur][scol + 8 + j][srow] = vb[j];
        }
        if (sb + 1 < nblocks) {
            // launch next block: async K -> buf[nxt], V -> registers
            const _Float16* kn = kbase + (size_t)(sb + 1) * 32 * H_;
            const _Float16* vn = vbase + (size_t)(sb + 1) * 32 * H_;
            const unsigned ldsk = (unsigned)(uintptr_t)&Kblk[nxt][srow][scol];
            asm volatile("global_load_async_to_lds_b128 %0, %1, off"
                         :: "v"(ldsk), "v"(kn) : "memory");
            asm volatile("global_load_async_to_lds_b128 %0, %1, off offset:16"
                         :: "v"(ldsk), "v"(kn) : "memory");
            va = *(const v8h*)(vn);
            vb = *(const v8h*)(vn + 8);
            if (sb + 2 < nblocks) {
                __builtin_prefetch(kn + 32 * H_, 0, 3);
                __builtin_prefetch(vn + 32 * H_, 0, 3);
            }
            // in-order async completion: <=2 outstanding => current pair done
            asm volatile("s_wait_asynccnt 2" ::: "memory");
        } else {
            asm volatile("s_wait_asynccnt 0" ::: "memory");
        }
        __syncthreads();   // Kblk[cur], VT[cur] ready for all waves

        // S = Q K^T (two 16-col tiles), causal mask in fp32 (scale is in q)
        float sval[2][8];
#pragma unroll
        for (int t = 0; t < 2; ++t) {
            v8f s_acc = {};
#pragma unroll
            for (int hs = 0; hs < 2; ++hs) {
                const _Float16* kr = &Kblk[cur][t * 16 + ln][hs * 32 + g * 8];
                v8h lo = *(const v8h*)kr;
                v8h hi = *(const v8h*)(kr + 16);
                v16h bf;
#pragma unroll
                for (int j = 0; j < 8; ++j) { bf[j] = lo[j]; bf[8 + j] = hi[j]; }
                s_acc = __builtin_amdgcn_wmma_f32_16x16x32_f16(
                            false, qf[hs], false, bf, (short)0, s_acc, false, false);
            }
#pragma unroll
            for (int r = 0; r < 8; ++r) {
                const int s_g = s0 + t * 16 + ln;
                const int q_g = wrow + r + 8 * g;
                sval[t][r] = (s_g <= q_g) ? s_acc[r] : -__builtin_inff();
            }
        }

        // online softmax: row reductions across the 16-lane stripe (wave32)
        float m_new[8], alpha[8];
#pragma unroll
        for (int r = 0; r < 8; ++r) {
            float mx = fmaxf(sval[0][r], sval[1][r]);
#pragma unroll
            for (int msk = 1; msk <= 8; msk <<= 1)
                mx = fmaxf(mx, __shfl_xor(mx, msk, 32));
            m_new[r] = fmaxf(m_i[r], mx);
            alpha[r] = __expf(m_i[r] - m_new[r]);
        }
#pragma unroll
        for (int r = 0; r < 8; ++r) {
            const float p0 = __expf(sval[0][r] - m_new[r]);
            const float p1 = __expf(sval[1][r] - m_new[r]);
            sval[0][r] = p0; sval[1][r] = p1;
            float s = p0 + p1;
#pragma unroll
            for (int msk = 1; msk <= 8; msk <<= 1)
                s += __shfl_xor(s, msk, 32);
            l_i[r] = l_i[r] * alpha[r] + s;
            m_i[r] = m_new[r];
        }
#pragma unroll
        for (int t = 0; t < 4; ++t)
#pragma unroll
            for (int r = 0; r < 8; ++r)
                o_acc[t][r] *= alpha[r];

        // P: D layout -> A layout via per-wave LDS scratch (same-wave DS order)
#pragma unroll
        for (int t = 0; t < 2; ++t)
#pragma unroll
            for (int r = 0; r < 8; ++r)
                Pblk[wave][r + 8 * g][t * 16 + ln] = (_Float16)sval[t][r];

        v16h pf;
        {
            const _Float16* pr = &Pblk[wave][ln][0];
            v8h lo = *(const v8h*)(pr + g * 8);
            v8h hi = *(const v8h*)(pr + 16 + g * 8);
#pragma unroll
            for (int j = 0; j < 8; ++j) { pf[j] = lo[j]; pf[8 + j] = hi[j]; }
        }

        // O += P V : V B-frags in table order from VT[cur][h][s]
#pragma unroll
        for (int ht = 0; ht < 4; ++ht) {
            const _Float16* vr = &VT[cur][ht * 16 + ln][g * 8];
            v8h lo = *(const v8h*)vr;
            v8h hi = *(const v8h*)(vr + 16);
            v16h vf;
#pragma unroll
            for (int j = 0; j < 8; ++j) { vf[j] = lo[j]; vf[8 + j] = hi[j]; }
            o_acc[ht] = __builtin_amdgcn_wmma_f32_16x16x32_f16(
                            false, pf, false, vf, (short)0, o_acc[ht], false, false);
        }
        // no trailing barrier: next iteration writes VT[nxt] and async-loads K into Kblk[cur]
    }

    // epilogue
#pragma unroll
    for (int ht = 0; ht < 4; ++ht)
#pragma unroll
        for (int r = 0; r < 8; ++r) {
            const int q_g = wrow + r + 8 * g;
            out[baseBT + (size_t)q_g * H_ + ht * 16 + ln] = o_acc[ht][r] / l_i[r];
        }
}

extern "C" void kernel_launch(void* const* d_in, const int* in_sizes, int n_in,
                              void* d_out, int out_size, void* d_ws, size_t ws_size,
                              hipStream_t stream) {
    const float* x  = (const float*)d_in[0];
    const float* Wq = (const float*)d_in[1];
    const float* Wk = (const float*)d_in[2];
    const float* Wv = (const float*)d_in[3];

    const size_t per = (size_t)B_ * T_ * H_;
    _Float16* qkv = (_Float16*)d_ws;           // [3][B*T][H] f16 = 6 MB
    _Float16* WT  = qkv + 3 * per;             // [3][H][C] f16 = 384 KB
    float* out = (float*)d_out;

    wt_kernel<<<(3 * H_ * C_) / 256, 256, 0, stream>>>(Wq, Wk, Wv, WT);
    qkv_proj_kernel<<<512, 256, 0, stream>>>(x, WT, qkv);   // 4096 wave tasks
    attn_kernel<<<B_ * (T_ / 64), 128, 0, stream>>>(qkv, qkv + per, qkv + 2 * per, out);
}
